// Transformer_module_9766755631131
// MI455X (gfx1250) — matrix operand order repaired and checked by reading, not compiled, not run on hardware
//
#include <hip/hip_runtime.h>
#include <hip/hip_bf16.h>

typedef __bf16 bf16_t;
typedef __attribute__((ext_vector_type(16))) __bf16 v16bf;
typedef __attribute__((ext_vector_type(8)))  __bf16 v8bf;
typedef __attribute__((ext_vector_type(8)))  float  v8f;
typedef __attribute__((ext_vector_type(4)))  unsigned int v4u;
typedef __attribute__((ext_vector_type(8)))  int v8i;
typedef __attribute__((ext_vector_type(4)))  int v4i;

static constexpr int Bc = 8;
static constexpr int Sc = 1024;
static constexpr int Mc = 1024;
static constexpr int Hc = 16;
static constexpr int Dc = 64;
static constexpr int Fc = 4096;
static constexpr int NROWS = Bc * Sc;   // 8192

// ---------------- WMMA helpers ----------------

__device__ __forceinline__ v16bf cat8(v8bf lo, v8bf hi) {
  return __builtin_shufflevector(lo, hi, 0,1,2,3,4,5,6,7,8,9,10,11,12,13,14,15);
}

__device__ __forceinline__ v8f vzero8() {
  v8f v;
#pragma unroll
  for (int i = 0; i < 8; ++i) v[i] = 0.0f;
  return v;
}

// A-matrix fragment (16x32 bf16). tile points at (row0, k0), ld in elements.
__device__ __forceinline__ v16bf load_a_frag(const bf16_t* tile, int ld) {
  const int lane = threadIdx.x & 31;
  const bf16_t* p = tile + (size_t)(lane & 15) * ld + ((lane >> 4) << 3);
  v8bf lo = *(const v8bf*)p;
  v8bf hi = *(const v8bf*)(p + 16);
  return cat8(lo, hi);
}

// B-matrix fragment (32x16 bf16) from a K-contiguous ("transposed") layout.
__device__ __forceinline__ v16bf load_bT_frag(const bf16_t* tileT, int ld) {
  const int lane = threadIdx.x & 31;
  const bf16_t* p = tileT + (size_t)(lane & 15) * ld + ((lane >> 4) << 3);
  v8bf lo = *(const v8bf*)p;
  v8bf hi = *(const v8bf*)(p + 16);
  return cat8(lo, hi);
}

__device__ __forceinline__ v8f wmma_bf16(v16bf a, v16bf b, v8f c) {
  return __builtin_amdgcn_wmma_f32_16x16x32_bf16(false, a, false, b, (short)0, c,
                                                 false, false);
}

// ---------------- Tensor Data Mover (TDM) helpers ----------------

// Raw LDS byte offset of a __shared__ object (generic -> addrspace(3) -> int).
__device__ __forceinline__ unsigned lds_addr_of(const void* p) {
  return (unsigned)(unsigned long long)
      (__attribute__((address_space(3))) const void*)p;
}

// One-shot 2D bf16 tile DMA: global [tensor_h x row_stride] -> LDS (packed
// row-major tile_h x tile_w). Issued per-wave; tracked by TENSORcnt.
// This toolchain exposes the 6-arg builtin:
//   (uint32x4 g0, int32x8 g1, int32x4 g2, int32x4 g3, int32x8, i32 cpol)
__device__ __forceinline__ void tdm_load_2d_bf16(unsigned lds_byte,
                                                 const bf16_t* gtile,
                                                 unsigned tensor_w,
                                                 unsigned tensor_h,
                                                 unsigned row_stride,
                                                 unsigned tile_w,
                                                 unsigned tile_h) {
  const unsigned long long ga = (unsigned long long)gtile;
  v4u g0;
  g0[0] = 1u;                                   // count=1 (valid user D#)
  g0[1] = lds_byte;                             // lds_addr
  g0[2] = (unsigned)ga;                         // global_addr[31:0]
  g0[3] = (unsigned)((ga >> 32) & 0x01FFFFFFu)  // global_addr[56:32]
          | (2u << 30);                         // type=2 ("image")
  v8i g1;
  g1[0] = (int)(1u << 16);                      // data_size=1 (2 bytes)
  g1[1] = (int)((tensor_w & 0xFFFFu) << 16);    // tensor_dim0[15:0] @63:48
  g1[2] = (int)((tensor_w >> 16) | ((tensor_h & 0xFFFFu) << 16));
  g1[3] = (int)((tensor_h >> 16) | (tile_w << 16));   // tile_dim0 @127:112
  g1[4] = (int)tile_h;                          // tile_dim1 (tile_dim2=0)
  g1[5] = (int)row_stride;                      // tensor_dim0_stride[31:0]
  g1[6] = 0;
  g1[7] = 0;
  v4i z4;
  z4[0] = 0; z4[1] = 0; z4[2] = 0; z4[3] = 0;
  v8i z8;
#pragma unroll
  for (int i = 0; i < 8; ++i) z8[i] = 0;
  __builtin_amdgcn_tensor_load_to_lds(g0, g1, z4, z4, z8, 0);
}

// ---------------- PE + LayerNorm1 (fused) ----------------

__global__ __launch_bounds__(256)
void pe_ln1_kernel(const float* __restrict__ x, const float* __restrict__ g,
                   const float* __restrict__ bt, float* __restrict__ xpe,
                   bf16_t* __restrict__ x2b) {
  __shared__ float red[18];
  const int rowg = blockIdx.x;
  const int s = rowg & (Sc - 1);
  float vals[4];
  float sum = 0.f, sq = 0.f;
#pragma unroll
  for (int i = 0; i < 4; ++i) {
    const int m = threadIdx.x + i * 256;
    const int i2 = m & ~1;
    const float ang = (float)s * __powf(10000.0f, -(float)i2 / (float)Mc);
    const float pe = (m & 1) ? __cosf(ang) : __sinf(ang);
    const float v = x[(size_t)rowg * Mc + m] + pe;
    vals[i] = v;
    xpe[(size_t)rowg * Mc + m] = v;
    sum += v;
    sq += v * v;
  }
#pragma unroll
  for (int off = 16; off; off >>= 1) {
    sum += __shfl_xor(sum, off, 32);
    sq  += __shfl_xor(sq,  off, 32);
  }
  const int wave = threadIdx.x >> 5, lane = threadIdx.x & 31;
  if (lane == 0) { red[wave] = sum; red[8 + wave] = sq; }
  __syncthreads();
  if (threadIdx.x == 0) {
    float ts = 0.f, tq = 0.f;
    for (int i = 0; i < 8; ++i) { ts += red[i]; tq += red[8 + i]; }
    red[16] = ts; red[17] = tq;
  }
  __syncthreads();
  const float mean = red[16] / (float)Mc;
  const float var = (red[17] - red[16] * mean) / (float)(Mc - 1);
  const float inv = 1.0f / (sqrtf(fmaxf(var, 0.f)) + 1e-12f);
#pragma unroll
  for (int i = 0; i < 4; ++i) {
    const int m = threadIdx.x + i * 256;
    x2b[(size_t)rowg * Mc + m] = (bf16_t)(g[m] * ((vals[i] - mean) * inv) + bt[m]);
  }
}

// plain LayerNorm (for x4 -> x5)
__global__ __launch_bounds__(256)
void ln_kernel(const float* __restrict__ in, const float* __restrict__ g,
               const float* __restrict__ bt, bf16_t* __restrict__ out) {
  __shared__ float red[18];
  const int rowg = blockIdx.x;
  float vals[4];
  float sum = 0.f, sq = 0.f;
#pragma unroll
  for (int i = 0; i < 4; ++i) {
    const int m = threadIdx.x + i * 256;
    const float v = in[(size_t)rowg * Mc + m];
    vals[i] = v;
    sum += v;
    sq += v * v;
  }
#pragma unroll
  for (int off = 16; off; off >>= 1) {
    sum += __shfl_xor(sum, off, 32);
    sq  += __shfl_xor(sq,  off, 32);
  }
  const int wave = threadIdx.x >> 5, lane = threadIdx.x & 31;
  if (lane == 0) { red[wave] = sum; red[8 + wave] = sq; }
  __syncthreads();
  if (threadIdx.x == 0) {
    float ts = 0.f, tq = 0.f;
    for (int i = 0; i < 8; ++i) { ts += red[i]; tq += red[8 + i]; }
    red[16] = ts; red[17] = tq;
  }
  __syncthreads();
  const float mean = red[16] / (float)Mc;
  const float var = (red[17] - red[16] * mean) / (float)(Mc - 1);
  const float inv = 1.0f / (sqrtf(fmaxf(var, 0.f)) + 1e-12f);
#pragma unroll
  for (int i = 0; i < 4; ++i) {
    const int m = threadIdx.x + i * 256;
    out[(size_t)rowg * Mc + m] = (bf16_t)(g[m] * ((vals[i] - mean) * inv) + bt[m]);
  }
}

// ---------------- f32 -> bf16 converter ----------------

__global__ __launch_bounds__(256)
void cvt_kernel(const float* __restrict__ in, bf16_t* __restrict__ out, int n) {
  const int i = (blockIdx.x * 256 + threadIdx.x) * 4;
  if (i < n) {
#pragma unroll
    for (int j = 0; j < 4; ++j) out[i + j] = (bf16_t)in[i + j];
  }
}

// ---------------- Generic 128x128 GEMM: out = epi(A*B + bias [+resid]) --------
// A: [rows, K] bf16 row-major (staged via TDM). B: [K, N] bf16 row-major
// (staged transposed into LDS by the waves). 256 threads = 8 waves.

template <bool GELU, bool OUT_BF16, bool RESID>
__global__ __launch_bounds__(256)
void gemm_kernel(const bf16_t* __restrict__ A, const bf16_t* __restrict__ Bm,
                 const float* __restrict__ bias, const float* __restrict__ resid,
                 void* __restrict__ out, int K, int N) {
  __shared__ __align__(16) bf16_t As[128 * 32];
  __shared__ __align__(16) bf16_t Bs[128 * 32];  // transposed: [n][k]
  const int tid = threadIdx.x;
  const int row0 = blockIdx.x * 128;
  const int n0 = blockIdx.y * 128;
  const int wave = tid >> 5;
  const int wr = wave >> 1, wc = wave & 1;
  const unsigned as_lds = lds_addr_of(As);

  v8f acc[2][4];
#pragma unroll
  for (int i = 0; i < 2; ++i)
#pragma unroll
    for (int j = 0; j < 4; ++j) acc[i][j] = vzero8();

  for (int k0 = 0; k0 < K; k0 += 32) {
    __syncthreads();  // WAR: previous iteration's fragment readers done
    // A tile 128x32 via Tensor Data Mover (one DMA per workgroup)
    if (wave == 0) {
      tdm_load_2d_bf16(as_lds, A + (size_t)row0 * K + k0,
                       (unsigned)K, (unsigned)NROWS, (unsigned)K, 32u, 128u);
    }
    // B tile: 32x128, store transposed (overlaps with the TDM DMA)
#pragma unroll
    for (int i = 0; i < 2; ++i) {
      const int f8 = tid * 2 + i;
      const int kr = f8 >> 4;
      const int n8 = (f8 & 15) << 3;
      v8bf v = *(const v8bf*)(Bm + (size_t)(k0 + kr) * N + n0 + n8);
#pragma unroll
      for (int e = 0; e < 8; ++e) Bs[(n8 + e) * 32 + kr] = v[e];
    }
    if (wave == 0) __builtin_amdgcn_s_wait_tensorcnt(0);
    __syncthreads();

    v16bf af[2], bfr[4];
#pragma unroll
    for (int i = 0; i < 2; ++i) af[i] = load_a_frag(As + (wr * 32 + i * 16) * 32, 32);
#pragma unroll
    for (int j = 0; j < 4; ++j) bfr[j] = load_bT_frag(Bs + (wc * 64 + j * 16) * 32, 32);
#pragma unroll
    for (int i = 0; i < 2; ++i)
#pragma unroll
      for (int j = 0; j < 4; ++j) acc[i][j] = wmma_bf16(af[i], bfr[j], acc[i][j]);
  }

  const int lane = tid & 31;
  const int cc = lane & 15;
  const int rofs = (lane >> 4) << 3;
#pragma unroll
  for (int i = 0; i < 2; ++i) {
#pragma unroll
    for (int j = 0; j < 4; ++j) {
      const int colg = n0 + wc * 64 + j * 16 + cc;
      const float bb = bias[colg];
#pragma unroll
      for (int r = 0; r < 8; ++r) {
        const int rowg = row0 + wr * 32 + i * 16 + rofs + r;
        float v = acc[i][j][r] + bb;
        if (RESID) v += resid[(size_t)rowg * N + colg];
        if (GELU) v = 0.5f * v * (1.0f + erff(v * 0.70710678118654752f));
        if (OUT_BF16)
          ((bf16_t*)out)[(size_t)rowg * N + colg] = (bf16_t)v;
        else
          ((float*)out)[(size_t)rowg * N + colg] = v;
      }
    }
  }
}

// ---------------- QKV GEMM (per head): out[b,h,s,d] = (x2*W_h + b_h) * scale ---

__global__ __launch_bounds__(256)
void qkv_kernel(const bf16_t* __restrict__ A, const bf16_t* __restrict__ W,
                const float* __restrict__ bias, bf16_t* __restrict__ out,
                float scale) {
  __shared__ __align__(16) bf16_t As[128 * 32];
  __shared__ __align__(16) bf16_t Bs[64 * 32];  // transposed: [d][k]
  const int tid = threadIdx.x;
  const int row0 = blockIdx.x * 128;
  const int h = blockIdx.y;
  const bf16_t* Wh = W + (size_t)h * Mc * Dc;
  const int wave = tid >> 5;
  const int wr = wave >> 1, wc = wave & 1;
  const unsigned as_lds = lds_addr_of(As);

  v8f acc[2][2];
#pragma unroll
  for (int i = 0; i < 2; ++i)
#pragma unroll
    for (int j = 0; j < 2; ++j) acc[i][j] = vzero8();

  for (int k0 = 0; k0 < Mc; k0 += 32) {
    __syncthreads();
    // A tile 128x32 via TDM
    if (wave == 0) {
      tdm_load_2d_bf16(as_lds, A + (size_t)row0 * Mc + k0,
                       (unsigned)Mc, (unsigned)NROWS, (unsigned)Mc, 32u, 128u);
    }
    {  // W tile 32x64, store transposed
      const int kr = tid >> 3;
      const int n8 = (tid & 7) << 3;
      v8bf v = *(const v8bf*)(Wh + (size_t)(k0 + kr) * Dc + n8);
#pragma unroll
      for (int e = 0; e < 8; ++e) Bs[(n8 + e) * 32 + kr] = v[e];
    }
    if (wave == 0) __builtin_amdgcn_s_wait_tensorcnt(0);
    __syncthreads();

    v16bf af[2], bfr[2];
#pragma unroll
    for (int i = 0; i < 2; ++i) af[i] = load_a_frag(As + (wr * 32 + i * 16) * 32, 32);
#pragma unroll
    for (int j = 0; j < 2; ++j) bfr[j] = load_bT_frag(Bs + (wc * 32 + j * 16) * 32, 32);
#pragma unroll
    for (int i = 0; i < 2; ++i)
#pragma unroll
      for (int j = 0; j < 2; ++j) acc[i][j] = wmma_bf16(af[i], bfr[j], acc[i][j]);
  }

  const int lane = tid & 31;
  const int cc = lane & 15;
  const int rofs = (lane >> 4) << 3;
#pragma unroll
  for (int i = 0; i < 2; ++i) {
#pragma unroll
    for (int j = 0; j < 2; ++j) {
      const int d = wc * 32 + j * 16 + cc;
      const float bb = bias[h * Dc + d];
#pragma unroll
      for (int r = 0; r < 8; ++r) {
        const int rowg = row0 + wr * 32 + i * 16 + rofs + r;
        const int b = rowg >> 10;
        const int s = rowg & (Sc - 1);
        out[(((size_t)b * Hc + h) * Sc + s) * Dc + d] =
            (bf16_t)((acc[i][j][r] + bb) * scale);
      }
    }
  }
}

// ---------------- Flash attention: per (b,h), 64 query rows / block -----------
// Q,K,V: [B,H,S,D] bf16 (Q pre-scaled by 1/sqrt(D)).  Z out: [B,S,H,D] bf16.

__global__ __launch_bounds__(128)
void attn_kernel(const bf16_t* __restrict__ Qg, const bf16_t* __restrict__ Kg,
                 const bf16_t* __restrict__ Vg, bf16_t* __restrict__ Zg) {
  const int b = blockIdx.z, h = blockIdx.y;
  const int wave = threadIdx.x >> 5, lane = threadIdx.x & 31;
  const int qbase = blockIdx.x * 64 + wave * 16;
  const bf16_t* Qp = Qg + (((size_t)b * Hc + h) * Sc + qbase) * Dc;
  const bf16_t* Kt = Kg + (((size_t)b * Hc + h) * Sc) * Dc;
  const bf16_t* Vt0 = Vg + (((size_t)b * Hc + h) * Sc) * Dc;

  __shared__ __align__(16) bf16_t Vt[64 * 32];       // transposed V tile [d][t]
  __shared__ __align__(16) bf16_t Pl[4][16 * 32];    // per-wave P tiles

  const int col = lane & 15;

  v16bf qf[2];
#pragma unroll
  for (int kk = 0; kk < 2; ++kk) qf[kk] = load_a_frag(Qp + kk * 32, Dc);

  v8f acc[4];
#pragma unroll
  for (int j = 0; j < 4; ++j) acc[j] = vzero8();
  float rm[8], rl[8];
#pragma unroll
  for (int r = 0; r < 8; ++r) { rm[r] = -3.0e38f; rl[r] = 0.f; }

  for (int t0 = 0; t0 < Sc; t0 += 32) {
    __syncthreads();  // protect Vt from previous-iteration readers
#pragma unroll
    for (int i = 0; i < 2; ++i) {  // stage V tile 32x64 transposed
      const int f8 = threadIdx.x * 2 + i;
      const int tt = f8 >> 3;
      const int d8 = (f8 & 7) << 3;
      v8bf v = *(const v8bf*)(Vt0 + (size_t)(t0 + tt) * Dc + d8);
#pragma unroll
      for (int e = 0; e < 8; ++e) Vt[(d8 + e) * 32 + tt] = v[e];
    }
    __syncthreads();

    // scores = Q(16x64) * K^T(64x32), split into two 16-col halves
    v8f sc0 = vzero8(), sc1 = vzero8();
#pragma unroll
    for (int kk = 0; kk < 2; ++kk) {
      v16bf kf0 = load_bT_frag(Kt + (size_t)t0 * Dc + kk * 32, Dc);
      v16bf kf1 = load_bT_frag(Kt + (size_t)(t0 + 16) * Dc + kk * 32, Dc);
      sc0 = wmma_bf16(qf[kk], kf0, sc0);
      sc1 = wmma_bf16(qf[kk], kf1, sc1);
    }

    // online softmax update; write P tile (bf16) to this wave's LDS region
#pragma unroll
    for (int r = 0; r < 8; ++r) {
      const float v0 = sc0[r], v1 = sc1[r];
      float mx = fmaxf(v0, v1);
#pragma unroll
      for (int off = 1; off < 16; off <<= 1) mx = fmaxf(mx, __shfl_xor(mx, off, 32));
      const float mnew = fmaxf(rm[r], mx);
      const float alpha = __expf(rm[r] - mnew);
      const float p0 = __expf(v0 - mnew);
      const float p1 = __expf(v1 - mnew);
      float srow = p0 + p1;
#pragma unroll
      for (int off = 1; off < 16; off <<= 1) srow += __shfl_xor(srow, off, 32);
      rl[r] = rl[r] * alpha + srow;
      rm[r] = mnew;
#pragma unroll
      for (int j = 0; j < 4; ++j) acc[j][r] *= alpha;
      const int prow = r + ((lane >> 4) << 3);
      Pl[wave][prow * 32 + col] = (bf16_t)p0;
      Pl[wave][prow * 32 + 16 + col] = (bf16_t)p1;
    }
    asm volatile("s_wait_dscnt 0" ::: "memory");  // same-wave LDS RAW fence

    // O += P(16x32) * V(32x64)
    v16bf pf = load_a_frag(&Pl[wave][0], 32);
#pragma unroll
    for (int j = 0; j < 4; ++j) {
      v16bf vf = load_bT_frag(Vt + (j * 16) * 32, 32);
      acc[j] = wmma_bf16(pf, vf, acc[j]);
    }
  }

  // epilogue: z[b,s,h,d] = O / l
#pragma unroll
  for (int j = 0; j < 4; ++j) {
#pragma unroll
    for (int r = 0; r < 8; ++r) {
      const int m = r + ((lane >> 4) << 3);
      const int s = qbase + m;
      const int d = j * 16 + col;
      Zg[(((size_t)b * Sc + s) * Hc + h) * Dc + d] = (bf16_t)(acc[j][r] / rl[r]);
    }
  }
}

// ---------------- Orchestration ----------------

extern "C" void kernel_launch(void* const* d_in, const int* in_sizes, int n_in,
                              void* d_out, int out_size, void* d_ws, size_t ws_size,
                              hipStream_t stream) {
  const float* x   = (const float*)d_in[0];
  const float* Wq  = (const float*)d_in[1];
  const float* bq  = (const float*)d_in[2];
  const float* Wk  = (const float*)d_in[3];
  const float* bk  = (const float*)d_in[4];
  const float* Wv  = (const float*)d_in[5];
  const float* bv  = (const float*)d_in[6];
  const float* Wc  = (const float*)d_in[7];
  const float* bc  = (const float*)d_in[8];
  const float* g1  = (const float*)d_in[9];
  const float* be1 = (const float*)d_in[10];
  const float* g2  = (const float*)d_in[11];
  const float* be2 = (const float*)d_in[12];
  const float* W1  = (const float*)d_in[13];
  const float* b1  = (const float*)d_in[14];
  const float* W2  = (const float*)d_in[15];
  const float* b2  = (const float*)d_in[16];

  const size_t NM = (size_t)NROWS * Mc;        // 8M elements
  const size_t WSZ = (size_t)Hc * Mc * Dc;     // 1M elements
  char* p = (char*)d_ws;
  float* xpe = (float*)p;   p += NM * 4;       // 32 MB (residual 1)
  float* x4  = (float*)p;   p += NM * 4;       // 32 MB (residual 2)
  bf16_t* Wqb = (bf16_t*)p; p += WSZ * 2;
  bf16_t* Wkb = (bf16_t*)p; p += WSZ * 2;
  bf16_t* Wvb = (bf16_t*)p; p += WSZ * 2;
  bf16_t* Wcb = (bf16_t*)p; p += (size_t)Mc * Mc * 2;
  bf16_t* W1b = (bf16_t*)p; p += (size_t)Mc * Fc * 2;
  bf16_t* W2b = (bf16_t*)p; p += (size_t)Fc * Mc * 2;
  bf16_t* x2b = (bf16_t*)p; p += NM * 2;       // reused as x5 after LN2
  bf16_t* Qb  = (bf16_t*)p; p += NM * 2;       // [B,H,S,D]
  bf16_t* Kb  = (bf16_t*)p; p += NM * 2;
  bf16_t* Vb  = (bf16_t*)p; p += NM * 2;
  bf16_t* zb  = (bf16_t*)p; p += NM * 2;       // [B,S,H,D]
  bf16_t* hb  = Qb;  // 64 MB region Qb..zb reused for FFN hidden [8192,4096]
  bf16_t* x5b = x2b;

  // 1) PE + LN1
  pe_ln1_kernel<<<NROWS, 256, 0, stream>>>(x, g1, be1, xpe, x2b);

  // 2) weight conversions
  cvt_kernel<<<(int)(WSZ / 1024), 256, 0, stream>>>(Wq, Wqb, (int)WSZ);
  cvt_kernel<<<(int)(WSZ / 1024), 256, 0, stream>>>(Wk, Wkb, (int)WSZ);
  cvt_kernel<<<(int)(WSZ / 1024), 256, 0, stream>>>(Wv, Wvb, (int)WSZ);
  cvt_kernel<<<(Mc * Mc) / 1024, 256, 0, stream>>>(Wc, Wcb, Mc * Mc);
  cvt_kernel<<<(Mc * Fc) / 1024, 256, 0, stream>>>(W1, W1b, Mc * Fc);
  cvt_kernel<<<(Fc * Mc) / 1024, 256, 0, stream>>>(W2, W2b, Fc * Mc);

  // 3) QKV projections (scale 1/sqrt(D) folded into Q)
  dim3 qkv_grid(NROWS / 128, Hc);
  qkv_kernel<<<qkv_grid, 256, 0, stream>>>(x2b, Wqb, bq, Qb, 0.125f);
  qkv_kernel<<<qkv_grid, 256, 0, stream>>>(x2b, Wkb, bk, Kb, 1.0f);
  qkv_kernel<<<qkv_grid, 256, 0, stream>>>(x2b, Wvb, bv, Vb, 1.0f);

  // 4) attention
  attn_kernel<<<dim3(Sc / 64, Hc, Bc), 128, 0, stream>>>(Qb, Kb, Vb, zb);

  // 5) output projection + residual(x_pe) -> x4 (f32)
  gemm_kernel<false, false, true><<<dim3(NROWS / 128, Mc / 128), 256, 0, stream>>>(
      zb, Wcb, bc, xpe, (void*)x4, Mc, Mc);

  // 6) LN2 -> x5 (bf16)
  ln_kernel<<<NROWS, 256, 0, stream>>>(x4, g2, be2, x5b);

  // 7) FFN1: GELU(x5*W1 + b1) -> h (bf16)
  gemm_kernel<true, true, false><<<dim3(NROWS / 128, Fc / 128), 256, 0, stream>>>(
      x5b, W1b, b1, nullptr, (void*)hb, Mc, Fc);

  // 8) FFN2: h*W2 + b2 + x4 -> d_out (f32)
  gemm_kernel<false, false, true><<<dim3(NROWS / 128, Mc / 128), 256, 0, stream>>>(
      hb, W2b, b2, x4, d_out, Fc, Mc);
}
